// GNNEncoder_83056077570937
// MI455X (gfx1250) — matrix operand order repaired and checked by reading, not compiled, not run on hardware
//
#include <hip/hip_runtime.h>
#include <hip/hip_bf16.h>

// GNN (hetero GINE) for MI455X / gfx1250.
// Live dataflow only:
//   xc0 = relu(x_customer@linc_W+b)
//   xm1 = leaky( GINE(xc0, c2m edges, layer1 weights[0]) )
//   out_c2 = GINE(xm1, m2c edges, layer2 weights[1])        -> d_out[0:N*H)
//   proj_c = relu(out_c2@pW1+pb1)@pW2+pb2                   -> d_out[N*H:2*N*H)
// Dense GEMMs: v_wmma_f32_16x16x32_bf16, A-stripes staged in LDS via the
// gfx1250 async global->LDS copy path (ASYNCcnt), fragments read with ds_load.

#define N_NODES 100000
#define EDGES   640000
#define H       128
#define NH      ((size_t)N_NODES * H)

typedef __attribute__((ext_vector_type(16))) __bf16 v16bf;
typedef __attribute__((ext_vector_type(8)))  float  v8f;
typedef int v4i_vs __attribute__((vector_size(16)));   // matches builtin's param type

__device__ __forceinline__ void atomic_add_f32(float* p, float v) {
    unsafeAtomicAdd(p, v);   // -> global_atomic_add_f32
}

// 16-byte global -> LDS copy on the CDNA5 async path when available.
// Builtin signature (from compiler diagnostic): global src is
// 'int __attribute__((vector_size(16))) __device__ *' (AS1, non-const).
__device__ __forceinline__ void copy16_to_lds(const float* g, float* l) {
#if defined(__has_builtin) && __has_builtin(__builtin_amdgcn_global_load_async_to_lds_b128)
    __builtin_amdgcn_global_load_async_to_lds_b128(
        (__attribute__((address_space(1))) v4i_vs*)g,
        (__attribute__((address_space(3))) v4i_vs*)l, 0, 0);
#else
    *(float4*)l = *(const float4*)g;
#endif
}
__device__ __forceinline__ void async_copy_drain() {
#if defined(__has_builtin) && __has_builtin(__builtin_amdgcn_global_load_async_to_lds_b128)
#if __has_builtin(__builtin_amdgcn_s_wait_asynccnt)
    __builtin_amdgcn_s_wait_asynccnt(0);
#else
    asm volatile("s_wait_asynccnt 0x0" ::: "memory");
#endif
#endif
}

// ---------------------------------------------------------------------------
// Pack a row-major f32 weight W[Ksrc x 128] into bf16 WMMA B-operand layout.
// Index = ((n_tile*Kchunks + c)*32 + lane)*16 + e; lane (h=lane>>4, n=lane&15),
// element e holds B[k][col] with k = c*32 + h*8 + e (e<8) or c*32 + 16 + h*8 + (e-8) (e>=8), col = n_tile*16 + n
// (zero-filled for k >= Ksrc).
// ---------------------------------------------------------------------------
__global__ void pack_w_kernel(const float* __restrict__ W, __bf16* __restrict__ out,
                              int Ksrc, int Kchunks) {
    int idx = blockIdx.x * blockDim.x + threadIdx.x;
    int total = 8 * Kchunks * 32 * 16;
    if (idx >= total) return;
    int e    = idx & 15;
    int lane = (idx >> 4) & 31;
    int c    = (idx >> 9) % Kchunks;
    int nt   = idx / (512 * Kchunks);
    int k    = c * 32 + ((lane >> 4) << 3) + (e & 7) + ((e >> 3) << 4);
    int col  = nt * 16 + (lane & 15);
    float v = (k < Ksrc) ? W[(size_t)k * H + col] : 0.0f;
    out[idx] = (__bf16)v;
}

// ---------------------------------------------------------------------------
// out[M x 128] = act(A[M x K] @ Wpacked + bias).  M multiple of 16.
// Block = 256 threads = 8 waves; block owns a 16-row stripe, wave w owns
// column tile w.  The 16xK f32 A-stripe is async-copied into LDS once per
// block (vs 8x redundant global reads), padded to stride K+4 floats so the
// ds_load_b64 fragment reads (16 lanes, stride-row) hit distinct bank pairs
// while keeping 16B alignment for the b128 copies.  ACT: 0=none,1=relu,2=leaky.
// ---------------------------------------------------------------------------
template<int K, int ACT>
__global__ __launch_bounds__(256) void gemm_wmma_kernel(
    const float* __restrict__ A, const __bf16* __restrict__ Wp,
    const float* __restrict__ bias, float* __restrict__ out) {
    constexpr int SPAD = K + 4;                 // floats per padded LDS row
    constexpr int SEGS = K / 4;                 // float4 segments per row
    __shared__ __align__(16) float Atile[16 * SPAD];

    int tile = blockIdx.x;
    int w    = threadIdx.x >> 5;     // column tile 0..7
    int lane = threadIdx.x & 31;
    int h    = lane >> 4;            // lane half
    int r    = lane & 15;            // row within tile (A) / col within tile (B,C)

    // cooperative async stage of the A stripe
    const float* gbase = A + (size_t)tile * 16 * K;
#pragma unroll
    for (int i = threadIdx.x; i < 16 * SEGS; i += 256) {
        int rr = i / SEGS, ss = i % SEGS;
        copy16_to_lds(gbase + (size_t)rr * K + ss * 4, &Atile[rr * SPAD + ss * 4]);
    }
    async_copy_drain();
    __syncthreads();

    const float*  arow = Atile + r * SPAD;
    const __bf16* wp   = Wp + (size_t)w * (K / 32) * 512 + (size_t)lane * 16;

    v8f acc = {};
#pragma unroll
    for (int c = 0; c < K / 32; ++c) {
        v16bf a;
        const float* ap = arow + c * 32 + h * 8;
#pragma unroll
        for (int v = 0; v < 4; ++v) {
            float2 lo = *(const float2*)(ap + 2 * v);        // K = c*32 + h*8 + 2v{,+1}
            float2 hi = *(const float2*)(ap + 16 + 2 * v);   // K = +16
            a[2 * v]     = (__bf16)lo.x;  a[2 * v + 1]     = (__bf16)lo.y;
            a[8 + 2 * v] = (__bf16)hi.x;  a[8 + 2 * v + 1] = (__bf16)hi.y;
        }
        v16bf b = *(const v16bf*)(wp + (size_t)c * 512);
        acc = __builtin_amdgcn_wmma_f32_16x16x32_bf16(false, a, false, b,
                                                      (short)0, acc, false, false);
    }

    int col = w * 16 + r;
    float bb = bias[col];
    float* orow = out + (size_t)(tile * 16) * H + col;
#pragma unroll
    for (int p = 0; p < 8; ++p) {          // C/D layout: VGPR p -> row p + h*8
        float v = acc[p] + bb;
        if (ACT == 1) v = fmaxf(v, 0.0f);
        if (ACT == 2) v = (v > 0.0f) ? v : 0.01f * v;
        orow[(size_t)(p + h * 8) * H] = v;
    }
}

// ---------------------------------------------------------------------------
// Fused GINE edge pass: per 16-edge tile, edge_emb = ea[16x16 pad32] @ eeW via
// one WMMA, then msg = relu(x_src[src[e]] + edge_emb + bias) scattered with
// f32 atomics into agg[dst[e]].  Block = 256 = 8 waves (one per col tile).
// Gathers/scatters are L2-resident (node matrix = 51 MB < 192 MB L2).
// ---------------------------------------------------------------------------
__global__ __launch_bounds__(256) void gine_edge_kernel(
    const float* __restrict__ ea, const int* __restrict__ src,
    const int* __restrict__ dst, const __bf16* __restrict__ Wp,
    const float* __restrict__ bias, const float* __restrict__ xsrc,
    float* __restrict__ agg) {
    int tile = blockIdx.x;
    int w    = threadIdx.x >> 5;
    int lane = threadIdx.x & 31;
    int h    = lane >> 4;
    int r    = lane & 15;

    // A fragment: edge row r, K<16 real, K>=16 zero pad.
    const float* erow = ea + (size_t)(tile * 16 + r) * 16 + h * 8;
    v16bf a;
#pragma unroll
    for (int v = 0; v < 4; ++v) {
        float2 q = *(const float2*)(erow + 2 * v);   // K = h*8 + 2v{,+1} < 16
        a[2 * v]     = (__bf16)q.x;      a[2 * v + 1]     = (__bf16)q.y;
        a[8 + 2 * v] = (__bf16)0.0f;     a[8 + 2 * v + 1] = (__bf16)0.0f;
    }
    v16bf b = *(const v16bf*)(Wp + ((size_t)w * 32 + lane) * 16);
    v8f acc = {};
    acc = __builtin_amdgcn_wmma_f32_16x16x32_bf16(false, a, false, b,
                                                  (short)0, acc, false, false);

    int col = w * 16 + r;
    float bb = bias[col];
#pragma unroll
    for (int p = 0; p < 8; ++p) {
        int e = tile * 16 + p + h * 8;
        int s = src[e];
        int d = dst[e];
        float msg = acc[p] + bb + xsrc[(size_t)s * H + col];  // gather
        msg = fmaxf(msg, 0.0f);
        atomic_add_f32(&agg[(size_t)d * H + col], msg);       // scatter-add
    }
}

// hagg = (1 + eps[epsIdx]) * x + hagg
__global__ void self_add_kernel(const float* __restrict__ x,
                                const float* __restrict__ eps, int epsIdx,
                                float* __restrict__ hagg, size_t n) {
    size_t i = (size_t)blockIdx.x * blockDim.x + threadIdx.x;
    if (i < n) hagg[i] = fmaf(1.0f + eps[epsIdx], x[i], hagg[i]);
}

// ---------------------------------------------------------------------------
extern "C" void kernel_launch(void* const* d_in, const int* in_sizes, int n_in,
                              void* d_out, int out_size, void* d_ws, size_t ws_size,
                              hipStream_t stream) {
    (void)in_sizes; (void)n_in; (void)out_size; (void)ws_size;
    const float* x_cust = (const float*)d_in[0];
    const int*   ei_c2m = (const int*)d_in[2];
    const int*   ei_m2c = (const int*)d_in[3];
    const float* ea_c2m = (const float*)d_in[4];
    const float* ea_m2c = (const float*)d_in[5];
    const float* linc_W = (const float*)d_in[6];
    const float* linc_b = (const float*)d_in[7];
    const float* eeW1   = (const float*)d_in[10];
    const float* eeb1   = (const float*)d_in[11];
    const float* m1W1   = (const float*)d_in[12];
    const float* m1b1   = (const float*)d_in[13];
    const float* m2W1   = (const float*)d_in[14];
    const float* m2b1   = (const float*)d_in[15];
    const float* eps1   = (const float*)d_in[16];
    const float* eeW2   = (const float*)d_in[17];
    const float* eeb2   = (const float*)d_in[18];
    const float* m1W2   = (const float*)d_in[19];
    const float* m1b2   = (const float*)d_in[20];
    const float* m2W2   = (const float*)d_in[21];
    const float* m2b2   = (const float*)d_in[22];
    const float* eps2   = (const float*)d_in[23];
    const float* pW1    = (const float*)d_in[24];
    const float* pb1    = (const float*)d_in[25];
    const float* pW2    = (const float*)d_in[26];
    const float* pb2    = (const float*)d_in[27];

    float* B0 = (float*)d_ws;        // node features / outputs
    float* B1 = B0 + NH;             // aggregation buffer
    __bf16* pk     = (__bf16*)(B1 + NH);
    __bf16* p_linc = pk;             // 64x128  -> 8192
    __bf16* p_ee1  = p_linc + 8192;  // 32x128  -> 4096
    __bf16* p_m11  = p_ee1 + 4096;   // 128x128 -> 16384
    __bf16* p_m21  = p_m11 + 16384;
    __bf16* p_ee2  = p_m21 + 16384;  // 4096
    __bf16* p_m12  = p_ee2 + 4096;
    __bf16* p_m22  = p_m12 + 16384;
    __bf16* p_p1   = p_m22 + 16384;
    __bf16* p_p2   = p_p1 + 16384;

    float* outC2 = (float*)d_out;    // first N*H of output
    float* projC = outC2 + NH;       // second N*H of output
    float* B2    = projC;            // reuse as MLP-hidden scratch until last step

    dim3 blk(256);
    auto pack = [&](const float* W, __bf16* dst, int Ksrc, int Kch) {
        int total = 8 * Kch * 32 * 16;
        pack_w_kernel<<<dim3((total + 255) / 256), blk, 0, stream>>>(W, dst, Ksrc, Kch);
    };
    pack(linc_W,            p_linc,  64, 2);
    pack(eeW1,              p_ee1,   16, 1);   // edge type 0 (c2m), layer 1
    pack(m1W1,              p_m11,  128, 4);
    pack(m2W1,              p_m21,  128, 4);
    pack(eeW2 + 16 * H,     p_ee2,   16, 1);   // edge type 1 (m2c), layer 2
    pack(m1W2 + H * H,      p_m12,  128, 4);
    pack(m2W2 + H * H,      p_m22,  128, 4);
    pack(pW1,               p_p1,   128, 4);
    pack(pW2,               p_p2,   128, 4);

    dim3 gN(N_NODES / 16);   // 6250
    dim3 gE(EDGES / 16);     // 40000
    int  gEW = (int)((NH + 255) / 256);

    // xc0 = relu(x_customer @ linc_W + linc_b)
    gemm_wmma_kernel<64, 1><<<gN, blk, 0, stream>>>(x_cust, p_linc, linc_b, B0);

    // --- layer 1, edge type c2m (produces merchant features xm1) ---
    (void)hipMemsetAsync(B1, 0, NH * sizeof(float), stream);
    gine_edge_kernel<<<gE, blk, 0, stream>>>(ea_c2m, ei_c2m, ei_c2m + EDGES,
                                             p_ee1, eeb1, B0, B1);
    self_add_kernel<<<gEW, blk, 0, stream>>>(B0, eps1, 0, B1, NH);
    gemm_wmma_kernel<128, 1><<<gN, blk, 0, stream>>>(B1, p_m11, m1b1, B2);
    gemm_wmma_kernel<128, 2><<<gN, blk, 0, stream>>>(B2, p_m21, m2b1, B0); // xm1 = leaky

    // --- layer 2, edge type m2c (produces out_c2) ---
    (void)hipMemsetAsync(B1, 0, NH * sizeof(float), stream);
    gine_edge_kernel<<<gE, blk, 0, stream>>>(ea_m2c, ei_m2c, ei_m2c + EDGES,
                                             p_ee2, eeb2 + H, B0, B1);
    self_add_kernel<<<gEW, blk, 0, stream>>>(B0, eps2, 1, B1, NH);
    gemm_wmma_kernel<128, 1><<<gN, blk, 0, stream>>>(B1, p_m12, m1b2 + H, B2);
    gemm_wmma_kernel<128, 0><<<gN, blk, 0, stream>>>(B2, p_m22, m2b2 + H, outC2);

    // --- projection head ---
    gemm_wmma_kernel<128, 1><<<gN, blk, 0, stream>>>(outC2, p_p1, pb1, B1);
    gemm_wmma_kernel<128, 0><<<gN, blk, 0, stream>>>(B1, p_p2, pb2, projC);
}
